// MCI_GRU_Cell_9182640079168
// MI455X (gfx1250) — compile-verified, matrix operand order checked by reading, not run on hardware
//
#include <hip/hip_runtime.h>

// ---------------------------------------------------------------------------
// CDNA5 (gfx1250) bf16 WMMA implementation of the MCI-GRU attention cell.
// All GEMMs run through v_wmma_f32_16x16x32_bf16 (wave32).
// All bf16 operands are stored pre-swizzled in the WMMA fragment layout so a
// fragment load is two b128 loads per lane (global or LDS). K loops are
// unrolled so the scheduler keeps many b128 loads in flight ahead of WMMA.
// f32->bf16 uses v_cvt_pk_bf16_f32 when the builtin exists.
// ---------------------------------------------------------------------------

typedef __attribute__((ext_vector_type(16))) __bf16 v16bf;
typedef __attribute__((ext_vector_type(2)))  __bf16 v2bf;
typedef __attribute__((ext_vector_type(8)))  float  v8f;

union FragU { v16bf v; unsigned int u[8]; };

__device__ __forceinline__ unsigned short f2bf_sw(float x) {
  unsigned int u = __float_as_uint(x);
  u += 0x7FFFu + ((u >> 16) & 1u);   // round-to-nearest-even
  return (unsigned short)(u >> 16);
}
// packed 2 x f32 -> 2 x bf16 in one dword (hardware cvt when available)
__device__ __forceinline__ unsigned int f2bf_pk(float lo, float hi) {
#if __has_builtin(__builtin_amdgcn_cvt_pk_bf16_f32)
  union { v2bf v; unsigned int u; } c;
  c.v = __builtin_amdgcn_cvt_pk_bf16_f32(lo, hi);
  return c.u;
#else
  return (unsigned int)f2bf_sw(lo) | ((unsigned int)f2bf_sw(hi) << 16);
#endif
}
__device__ __forceinline__ unsigned short f2bf1(float x) {
#if __has_builtin(__builtin_amdgcn_cvt_pk_bf16_f32)
  return (unsigned short)f2bf_pk(x, 0.0f);
#else
  return f2bf_sw(x);
#endif
}

// ---- swizzled layouts ------------------------------------------------------
// ASW(M,K): blocks of 16 rows x 32 K; block = 32 lanes x 32B (512 ushorts).
//           lane = (m&15) | (half<<4); dword j holds K pair frag_k(j,half).
// BSW(K,N): blocks of 32 K x 16 cols; lane = (n&15) | (half<<4); same K pairs.
// Fragment load = 2 x b128 from (base + bi*512 ushorts + lane*16 ushorts).

// ushort-element offset of (m,k) inside ASW(M,K)
__device__ __forceinline__ int asw_idx(int m, int k, int K) {
  int kk = k & 31;
  int half = (kk >> 3) & 1;
  int j = ((kk >> 4) << 2) | ((kk >> 1) & 3);
  int pair = kk & 1;
  int lane = (m & 15) | (half << 4);
  int bi = (m >> 4) * (K >> 5) + (k >> 5);
  return (bi << 9) + (lane << 4) + (j << 1) + pair;
}
// ushort-element offset of (k,n) inside BSW(K,N)
__device__ __forceinline__ int bsw_idx(int k, int n, int N) {
  int kk = k & 31;
  int half = (kk >> 3) & 1;
  int j = ((kk >> 4) << 2) | ((kk >> 1) & 3);
  int pair = kk & 1;
  int lane = (n & 15) | (half << 4);
  int bi = (k >> 5) * (N >> 4) + (n >> 4);
  return (bi << 9) + (lane << 4) + (j << 1) + pair;
}

// Linearized C-tile store bases: for a fixed target column gc (ASW) or fixed
// target K (BSW) the swizzled address is base + ((half<<3)+i)*16 over the 8
// accumulator rows of this lane's C column.
__device__ __forceinline__ int asw_store_base(int rowtile16, int gc, int K) {
  int kk = gc & 31;
  int halfT = (kk >> 3) & 1;
  int jj = ((kk >> 4) << 2) | ((kk >> 1) & 3);
  int pair = kk & 1;
  int bi = (rowtile16 >> 4) * (K >> 5) + (gc >> 5);
  return (bi << 9) + (halfT << 8) + (jj << 1) + pair;
}
__device__ __forceinline__ int bsw_store_base(int k, int ntile16, int N) {
  int kk = k & 31;
  int halfT = (kk >> 3) & 1;
  int jj = ((kk >> 4) << 2) | ((kk >> 1) & 3);
  int pair = kk & 1;
  int bi = (k >> 5) * (N >> 4) + (ntile16 >> 4);
  return (bi << 9) + (halfT << 8) + (jj << 1) + pair;
}

// Fragment load from swizzled storage (global or LDS): 2 x b128 per lane.
__device__ __forceinline__ v16bf load_frag(const unsigned short* base, int bi) {
  int lane = threadIdx.x & 31;
  const uint4* p = reinterpret_cast<const uint4*>(base + ((size_t)bi << 9)) + (lane << 1);
  uint4 lo = p[0];
  uint4 hi = p[1];
  FragU f;
  f.u[0] = lo.x; f.u[1] = lo.y; f.u[2] = lo.z; f.u[3] = lo.w;
  f.u[4] = hi.x; f.u[5] = hi.y; f.u[6] = hi.z; f.u[7] = hi.w;
  return f.v;
}

__device__ __forceinline__ v8f wmma_bf16(v16bf a, v16bf b, v8f c) {
  return __builtin_amdgcn_wmma_f32_16x16x32_bf16(false, a, false, b,
                                                 (short)0, c, false, false);
}

__device__ __forceinline__ float sigmoidf_(float x) {
  return 1.0f / (1.0f + __expf(-x));
}

// ---------------------------------------------------------------------------
// Weight convert: row-major f32 [K x N] -> BSW(K,N) bf16 (K-pairs packed).
// ---------------------------------------------------------------------------
__global__ void k_w_to_bsw(const float* __restrict__ src,
                           unsigned short* __restrict__ dst, int K, int N) {
  int p = blockIdx.x * blockDim.x + threadIdx.x;
  if (p < (K >> 1) * N) {
    int k = (p / N) << 1, n = p - (p / N) * N;
    unsigned int* d32 = (unsigned int*)dst;
    d32[bsw_idx(k, n, N) >> 1] = f2bf_pk(src[(size_t)k * N + n],
                                         src[(size_t)(k + 1) * N + n]);
  }
}

// Row-major f32 [M x K] -> ASW(M,K) bf16 (K-pairs packed). Used for h0.
__global__ void k_f32_to_asw(const float* __restrict__ src,
                             unsigned short* __restrict__ dst, int M, int K) {
  int p = blockIdx.x * blockDim.x + threadIdx.x;
  if (p < M * (K >> 1)) {
    int m = p / (K >> 1);
    int k = (p - m * (K >> 1)) << 1;
    unsigned int* d32 = (unsigned int*)dst;
    d32[asw_idx(m, k, K) >> 1] = f2bf_pk(src[(size_t)m * K + k],
                                         src[(size_t)m * K + k + 1]);
  }
}

// x [B,T,D] f32 -> time-major xbf: for each t, ASW(512,128) bf16.
__global__ void k_x_to_asw(const float* __restrict__ x,
                           unsigned short* __restrict__ xbf) {
  int p = blockIdx.x * blockDim.x + threadIdx.x;  // pair index, D/2=64 pairs
  int d = (p & 63) << 1;
  int t = (p >> 6) & 127;
  int b = p >> 13;
  const float* s = x + ((size_t)b * 128 + t) * 128 + d;
  unsigned int* dst = (unsigned int*)(xbf + (size_t)t * 512 * 128);
  dst[asw_idx(b, d, 128) >> 1] = f2bf_pk(s[0], s[1]);
}

// ---------------------------------------------------------------------------
// XT[t] = x[:,t,:] @ W_input, time-major ASW: XT + t*B*U in ASW(512,256).
// Grid: (B/32, T). WG = 8 waves, tile 32x256, wave tile 16x64.
// ---------------------------------------------------------------------------
__global__ void k_xt(const unsigned short* __restrict__ xbf,
                     const unsigned short* __restrict__ WiB,
                     unsigned short* __restrict__ XT) {
  int t = blockIdx.y;
  int wid = threadIdx.x >> 5, lane = threadIdx.x & 31;
  int row = blockIdx.x * 32 + (wid >> 2) * 16;   // batch row
  int col = (wid & 3) * 64;
  const unsigned short* A = xbf + (size_t)t * 512 * 128;   // ASW(512,128)
  unsigned short* dst = XT + (size_t)t * 512 * 256;
  v8f acc[4] = {};
#pragma unroll
  for (int kb = 0; kb < 128; kb += 32) {
    v16bf a = load_frag(A, (row >> 4) * 4 + (kb >> 5));
#pragma unroll
    for (int c = 0; c < 4; ++c) {
      v16bf b = load_frag(WiB, (kb >> 5) * 16 + ((col + c * 16) >> 4));
      acc[c] = wmma_bf16(a, b, acc[c]);
    }
  }
  int n = lane & 15, half = lane >> 4;
#pragma unroll
  for (int c = 0; c < 4; ++c) {
    int base = asw_store_base(row, col + c * 16 + n, 256);
#pragma unroll
    for (int i = 0; i < 8; ++i)
      dst[base + (((half << 3) + i) << 4)] = f2bf1(acc[c][i]);
  }
}

// ---------------------------------------------------------------------------
// Fused QKV: one pass over h shares every A fragment across the 3 GEMMs.
//   q = (h@W_q)/16 -> ASW(512,256)
//   kT = (h@W_k)^T -> BSW(256,512)
//   v  = h@W_v     -> BSW(512,256)
// M=512, N=256, K=256.  WG tile 32x256, wave tile 16x64, 12 WMMA per K-step.
// ---------------------------------------------------------------------------
__global__ void k_qkv(const unsigned short* __restrict__ h,
                      const unsigned short* __restrict__ Wq,
                      const unsigned short* __restrict__ Wk,
                      const unsigned short* __restrict__ Wv,
                      unsigned short* __restrict__ q,
                      unsigned short* __restrict__ kT,
                      unsigned short* __restrict__ v) {
  int wid = threadIdx.x >> 5, lane = threadIdx.x & 31;
  int row = blockIdx.x * 32 + (wid >> 2) * 16;
  int col = (wid & 3) * 64;
  v8f qacc[4] = {};
  v8f kacc[4] = {};
  v8f vacc[4] = {};
#pragma unroll
  for (int kb = 0; kb < 256; kb += 32) {
    v16bf a = load_frag(h, (row >> 4) * 8 + (kb >> 5));
#pragma unroll
    for (int c = 0; c < 4; ++c) {
      int wbi = (kb >> 5) * 16 + ((col + c * 16) >> 4);
      qacc[c] = wmma_bf16(a, load_frag(Wq, wbi), qacc[c]);
      kacc[c] = wmma_bf16(a, load_frag(Wk, wbi), kacc[c]);
      vacc[c] = wmma_bf16(a, load_frag(Wv, wbi), vacc[c]);
    }
  }
  int n = lane & 15, half = lane >> 4;
#pragma unroll
  for (int c = 0; c < 4; ++c) {
    int gc = col + c * 16 + n;
    int qbase = asw_store_base(row, gc, 256);       // q: A operand
    int kbase = bsw_store_base(gc, row, 512);       // kT: (k = unit, n = batch)
#pragma unroll
    for (int i = 0; i < 8; ++i) {
      int off = ((half << 3) + i) << 4;
      q[qbase + off] = f2bf1(qacc[c][i] * 0.0625f); // 1/sqrt(256)
      kT[kbase + off] = f2bf1(kacc[c][i]);
      v[bsw_idx(row + (half << 3) + i, gc, 256)] = f2bf1(vacc[c][i]);
    }
  }
}

// ---------------------------------------------------------------------------
// Fused scores -> softmax -> context for a 16-row block.
//   scores[16,512] = q[16,256] @ kT     (q pre-scaled)
//   ctx[16,256]    = softmax(scores) @ v
// ---------------------------------------------------------------------------
__global__ void k_attn(const unsigned short* __restrict__ q,
                       const unsigned short* __restrict__ kT,
                       const unsigned short* __restrict__ v,
                       float* __restrict__ ctxf,
                       unsigned short* __restrict__ ctxb) {
  __shared__ __attribute__((aligned(16))) float sc[16 * 512];           // 32 KB
  __shared__ __attribute__((aligned(16))) unsigned short wA[16 * 512];  // 16 KB
  __shared__ float red[16 * 16];
  __shared__ float rowstat[16];
  int rowbase = blockIdx.x * 16;
  int wid = threadIdx.x >> 5, lane = threadIdx.x & 31;
  int n = lane & 15, half = lane >> 4;

  { // ---- scores: 8 waves each own 64 columns of the 512 ----
    int col = wid * 64;
    v8f acc[4] = {};
#pragma unroll
    for (int kb = 0; kb < 256; kb += 32) {
      v16bf a = load_frag(q, (rowbase >> 4) * 8 + (kb >> 5));
#pragma unroll
      for (int c = 0; c < 4; ++c) {
        v16bf b = load_frag(kT, (kb >> 5) * 32 + ((col + c * 16) >> 4));
        acc[c] = wmma_bf16(a, b, acc[c]);
      }
    }
#pragma unroll
    for (int c = 0; c < 4; ++c)
#pragma unroll
      for (int i = 0; i < 8; ++i)
        sc[((half << 3) + i) * 512 + col + c * 16 + n] = acc[c][i];
  }
  __syncthreads();

  { // ---- row softmax over 512 cols: 16 threads/row, 32 cols each ----
    int r = threadIdx.x >> 4;
    int s = threadIdx.x & 15;
    float m = -3.4e38f;
    for (int c = s * 32; c < s * 32 + 32; ++c) m = fmaxf(m, sc[r * 512 + c]);
    red[r * 16 + s] = m;
    __syncthreads();
    if (s == 0) {
      float mm = red[r * 16];
      for (int j = 1; j < 16; ++j) mm = fmaxf(mm, red[r * 16 + j]);
      rowstat[r] = mm;
    }
    __syncthreads();
    float mm = rowstat[r];
    float sum = 0.f;
    for (int c = s * 32; c < s * 32 + 32; ++c) {
      float e = __expf(sc[r * 512 + c] - mm);
      sc[r * 512 + c] = e;
      sum += e;
    }
    red[r * 16 + s] = sum;
    __syncthreads();
    if (s == 0) {
      float ss = 0.f;
      for (int j = 0; j < 16; ++j) ss += red[r * 16 + j];
      rowstat[r] = 1.0f / ss;
    }
    __syncthreads();
    float inv = rowstat[r];
    // normalized weights -> A-swizzled bf16 LDS, packed dword stores
    unsigned int* wA32 = (unsigned int*)wA;
    for (int c = s * 32; c < s * 32 + 32; c += 2)
      wA32[asw_idx(r, c, 512) >> 1] =
          f2bf_pk(sc[r * 512 + c] * inv, sc[r * 512 + c + 1] * inv);
  }
  __syncthreads();

  { // ---- context: w(LDS, ASW) @ v ; 8 waves each own 32 columns ----
    int col = wid * 32;
    v8f acc[2] = {};
#pragma unroll 8
    for (int kb = 0; kb < 512; kb += 32) {
      v16bf a = load_frag(wA, kb >> 5);            // ASW(16,512): bi = kblock
#pragma unroll
      for (int c = 0; c < 2; ++c) {
        v16bf b = load_frag(v, (kb >> 5) * 16 + ((col + c * 16) >> 4));
        acc[c] = wmma_bf16(a, b, acc[c]);
      }
    }
#pragma unroll
    for (int c = 0; c < 2; ++c) {
      int base = asw_store_base(rowbase, col + c * 16 + n, 256);
#pragma unroll
      for (int i = 0; i < 8; ++i) {
        int gr = rowbase + (half << 3) + i;
        int gc = col + c * 16 + n;
        float val = acc[c][i];
        ctxf[(long)gr * 256 + gc] = val;
        ctxb[base + (((half << 3) + i) << 4)] = f2bf1(val);
      }
    }
  }
}

// ---------------------------------------------------------------------------
// Fused gates + state update for a 32-row block.
//   z  = sigmoid(xt@Wz + ctx@Uz + bz)
//   r  = sigmoid(xt@Wr + ctx@Ur + br);  rc = r*ctx (LDS, ASW bf16)
//   h~ = tanh(xt@Wh + rc@Uh + bh)
//   h  = (1-z)*ctx + z*h~   -> out[:,t,:] f32 and h (ASW bf16) for next step.
// ---------------------------------------------------------------------------
__global__ void k_gates(const unsigned short* __restrict__ XTt,   // ASW(512,256)
                        const unsigned short* __restrict__ ctxb,  // ASW(512,256)
                        const float* __restrict__ ctxf,
                        const unsigned short* __restrict__ Wz,
                        const unsigned short* __restrict__ Uz,
                        const float* __restrict__ bz,
                        const unsigned short* __restrict__ Wr,
                        const unsigned short* __restrict__ Ur,
                        const float* __restrict__ br,
                        const unsigned short* __restrict__ Wh,
                        const unsigned short* __restrict__ Uh,
                        const float* __restrict__ bh,
                        float* __restrict__ hout, long hout_ld,
                        unsigned short* __restrict__ hbf) {
  __shared__ __attribute__((aligned(16))) unsigned short rc[32 * 256]; // ASW(32,256)
  int rowbase = blockIdx.x * 32;
  int wid = threadIdx.x >> 5, lane = threadIdx.x & 31;
  int lrow = (wid >> 2) * 16;
  int col = (wid & 3) * 64;
  int n = lane & 15, half = lane >> 4;
  int row = rowbase + lrow;

  // ---- z and r gates (shared A fragments) ----
  v8f zacc[4] = {};
  v8f racc[4] = {};
#pragma unroll
  for (int kb = 0; kb < 256; kb += 32) {
    v16bf ax = load_frag(XTt, (row >> 4) * 8 + (kb >> 5));
    v16bf ac = load_frag(ctxb, (row >> 4) * 8 + (kb >> 5));
#pragma unroll
    for (int c = 0; c < 4; ++c) {
      int wbi = (kb >> 5) * 16 + ((col + c * 16) >> 4);
      zacc[c] = wmma_bf16(ax, load_frag(Wz, wbi), zacc[c]);
      zacc[c] = wmma_bf16(ac, load_frag(Uz, wbi), zacc[c]);
      racc[c] = wmma_bf16(ax, load_frag(Wr, wbi), racc[c]);
      racc[c] = wmma_bf16(ac, load_frag(Ur, wbi), racc[c]);
    }
  }
  float zval[4][8];
#pragma unroll
  for (int c = 0; c < 4; ++c) {
    int base = asw_store_base(lrow, col + c * 16 + n, 256);   // rc is local ASW(32,256)
#pragma unroll
    for (int i = 0; i < 8; ++i) {
      int lr = lrow + (half << 3) + i;
      int gc = col + c * 16 + n;
      zval[c][i] = sigmoidf_(zacc[c][i] + bz[gc]);
      float rr = sigmoidf_(racc[c][i] + br[gc]);
      float cx = ctxf[(long)(rowbase + lr) * 256 + gc];
      rc[base + (((half << 3) + i) << 4)] = f2bf1(rr * cx);
    }
  }
  __syncthreads();

  // ---- candidate + update ----
  v8f hacc[4] = {};
#pragma unroll
  for (int kb = 0; kb < 256; kb += 32) {
    v16bf ax = load_frag(XTt, (row >> 4) * 8 + (kb >> 5));
    v16bf ar = load_frag(rc, (lrow >> 4) * 8 + (kb >> 5));
#pragma unroll
    for (int c = 0; c < 4; ++c) {
      int wbi = (kb >> 5) * 16 + ((col + c * 16) >> 4);
      hacc[c] = wmma_bf16(ax, load_frag(Wh, wbi), hacc[c]);
      hacc[c] = wmma_bf16(ar, load_frag(Uh, wbi), hacc[c]);
    }
  }
#pragma unroll
  for (int c = 0; c < 4; ++c) {
    int base = asw_store_base(row, col + c * 16 + n, 256);
#pragma unroll
    for (int i = 0; i < 8; ++i) {
      int lr = lrow + (half << 3) + i;
      int gc = col + c * 16 + n;
      float ht = tanhf(hacc[c][i] + bh[gc]);
      float cx = ctxf[(long)(rowbase + lr) * 256 + gc];
      float z = zval[c][i];
      float h = (1.0f - z) * cx + z * ht;
      hout[(long)(rowbase + lr) * hout_ld + gc] = h;
      hbf[base + (((half << 3) + i) << 4)] = f2bf1(h);
    }
  }
}

// ---------------------------------------------------------------------------
// Host launch
// ---------------------------------------------------------------------------
extern "C" void kernel_launch(void* const* d_in, const int* in_sizes, int n_in,
                              void* d_out, int out_size, void* d_ws, size_t ws_size,
                              hipStream_t stream) {
  (void)in_sizes; (void)n_in; (void)out_size; (void)ws_size;
  const int B = 512, T = 128, D = 128, U = 256;

  const float* x  = (const float*)d_in[0];
  const float* h0 = (const float*)d_in[1];
  const float* Wi = (const float*)d_in[2];
  const float* Wz = (const float*)d_in[3];
  const float* Uz = (const float*)d_in[4];
  const float* bz = (const float*)d_in[5];
  const float* Wr = (const float*)d_in[6];
  const float* Ur = (const float*)d_in[7];
  const float* br = (const float*)d_in[8];
  const float* Wh = (const float*)d_in[9];
  const float* Uh = (const float*)d_in[10];
  const float* bh = (const float*)d_in[11];
  const float* Wq = (const float*)d_in[12];
  const float* Wk = (const float*)d_in[13];
  const float* Wv = (const float*)d_in[14];
  float* out = (float*)d_out;

  char* ws = (char*)d_ws;
  size_t off = 0;
  auto take = [&](size_t bytes) -> char* {
    char* p = ws + off;
    off = (off + bytes + 255) & ~(size_t)255;
    return p;
  };
  unsigned short* xbf = (unsigned short*)take((size_t)B * T * D * 2); // time-major ASW
  unsigned short* XT  = (unsigned short*)take((size_t)B * T * U * 2); // time-major ASW
  unsigned short* WiB = (unsigned short*)take((size_t)D * U * 2);
  unsigned short* WzB = (unsigned short*)take((size_t)U * U * 2);
  unsigned short* UzB = (unsigned short*)take((size_t)U * U * 2);
  unsigned short* WrB = (unsigned short*)take((size_t)U * U * 2);
  unsigned short* UrB = (unsigned short*)take((size_t)U * U * 2);
  unsigned short* WhB = (unsigned short*)take((size_t)U * U * 2);
  unsigned short* UhB = (unsigned short*)take((size_t)U * U * 2);
  unsigned short* WqB = (unsigned short*)take((size_t)U * U * 2);
  unsigned short* WkB = (unsigned short*)take((size_t)U * U * 2);
  unsigned short* WvB = (unsigned short*)take((size_t)U * U * 2);
  unsigned short* hB  = (unsigned short*)take((size_t)B * U * 2);
  unsigned short* qB  = (unsigned short*)take((size_t)B * U * 2);
  unsigned short* kTB = (unsigned short*)take((size_t)U * B * 2);
  unsigned short* vB  = (unsigned short*)take((size_t)B * U * 2);
  float*          ctxF = (float*)take((size_t)B * U * 4);
  unsigned short* ctxB = (unsigned short*)take((size_t)B * U * 2);

  auto cvtB = [&](const float* s, unsigned short* d, int K, int N) {
    k_w_to_bsw<<<((K / 2) * N + 255) / 256, 256, 0, stream>>>(s, d, K, N);
  };
  cvtB(Wi, WiB, D, U);
  cvtB(Wz, WzB, U, U); cvtB(Uz, UzB, U, U);
  cvtB(Wr, WrB, U, U); cvtB(Ur, UrB, U, U);
  cvtB(Wh, WhB, U, U); cvtB(Uh, UhB, U, U);
  cvtB(Wq, WqB, U, U); cvtB(Wk, WkB, U, U); cvtB(Wv, WvB, U, U);
  k_f32_to_asw<<<(B * (U / 2) + 255) / 256, 256, 0, stream>>>(h0, hB, B, U);
  k_x_to_asw<<<(B * T * D / 2) / 256, 256, 0, stream>>>(x, xbf);

  // Hoisted input projection for all timesteps (time-major ASW)
  k_xt<<<dim3(B / 32, T), 256, 0, stream>>>(xbf, WiB, XT);

  for (int t = 0; t < T; ++t) {
    k_qkv<<<dim3(B / 32), 256, 0, stream>>>(hB, WqB, WkB, WvB, qB, kTB, vB);
    k_attn<<<dim3(B / 16), 256, 0, stream>>>(qB, kTB, vB, ctxF, ctxB);
    k_gates<<<dim3(B / 32), 256, 0, stream>>>(
        XT + (size_t)t * B * U, ctxB, ctxF,
        WzB, UzB, bz, WrB, UrB, br, WhB, UhB, bh,
        out + (size_t)t * U, (long)T * U, hB);
  }
}
